// ManifoldConstrainedHyperConnection_68985764708573
// MI455X (gfx1250) — matrix operand order repaired and checked by reading, not compiled, not run on hardware
//
#include <hip/hip_runtime.h>
#include <hip/hip_bf16.h>

// ---------------------------------------------------------------------------
// ManifoldConstrainedHyperConnection for MI455X (gfx1250, wave32, WMMA)
//   SEQ=8192, N=4, D=1024, ND=4096
// ---------------------------------------------------------------------------

#define SEQ   8192
#define NSTR  4
#define DDIM  1024
#define NDF   4096
#define EPSF  1e-5f
#define SKIT  20

typedef __attribute__((ext_vector_type(16))) __bf16 v16bf;
typedef __attribute__((ext_vector_type(8)))  float  v8f;
typedef int v4i __attribute__((vector_size(16)));   // matches builtin param type

#if __has_builtin(__builtin_amdgcn_global_load_async_to_lds_b128) && \
    __has_builtin(__builtin_amdgcn_s_wait_asynccnt)
#define HAVE_ASYNC_LDS 1
#else
#define HAVE_ASYNC_LDS 0
#endif

__device__ __forceinline__ unsigned short f2bf(float f) {
    unsigned u = __float_as_uint(f);
    unsigned r = (u + 0x7FFFu + ((u >> 16) & 1u)) >> 16;   // RNE, ignore NaN edge
    return (unsigned short)r;
}

__device__ __forceinline__ float waveReduce(float v) {
    // wave32 reduction
    for (int o = 16; o > 0; o >>= 1) v += __shfl_down(v, o, 32);
    return v;
}

// ---------------------------------------------------------------------------
// Kernel 1: per-token RMSNorm + 24 skinny dots + gates + Sinkhorn +
//           layer_input (bf16) to workspace.  One block (256 thr) per token.
// ---------------------------------------------------------------------------
__global__ __launch_bounds__(256) void prep_kernel(
    const float* __restrict__ x,        // (SEQ, 4, 1024)
    const float* __restrict__ w_pre,    // (4, 4096)
    const float* __restrict__ w_post,   // (4, 4096)
    const float* __restrict__ w_res,    // (16, 4096)
    const float* __restrict__ b_pre,    // (4)
    const float* __restrict__ b_post,   // (4)
    const float* __restrict__ b_res,    // (16)
    const float* __restrict__ alpha_pre,
    const float* __restrict__ alpha_post,
    const float* __restrict__ alpha_res,
    const float* __restrict__ rms_w,    // (4096)
    const float* __restrict__ rms_b,    // (4096)
    unsigned short* __restrict__ A_bf,  // (8192, 1024) bf16 layer_input
    float* __restrict__ hpost_ws,       // (8192, 4)
    float* __restrict__ hres_ws)        // (8192, 16)
{
    const int s    = blockIdx.x;
    const int tid  = threadIdx.x;
    const int lane = tid & 31;
    const int wid  = tid >> 5;

    __shared__ float wred[8];
    __shared__ float dred[24][8];
    __shared__ float sh_inv;
    __shared__ float sh_dot[24];
    __shared__ float sh_hpre[4];

    const float* xrow = x + ((size_t)s << 12);       // 4096 floats
    const int k0 = tid * 16;

    // ---- load this thread's 16 x values -------------------------------
    float xr[16];
    {
        const float4* xv = (const float4*)(xrow + k0);
        float4 a0 = xv[0], a1 = xv[1], a2 = xv[2], a3 = xv[3];
        xr[0]=a0.x; xr[1]=a0.y; xr[2]=a0.z;  xr[3]=a0.w;
        xr[4]=a1.x; xr[5]=a1.y; xr[6]=a1.z;  xr[7]=a1.w;
        xr[8]=a2.x; xr[9]=a2.y; xr[10]=a2.z; xr[11]=a2.w;
        xr[12]=a3.x;xr[13]=a3.y;xr[14]=a3.z; xr[15]=a3.w;
    }

    // ---- sum of squares ----------------------------------------------
    float ss = 0.f;
    #pragma unroll
    for (int k = 0; k < 16; ++k) ss += xr[k] * xr[k];
    ss = waveReduce(ss);
    if (lane == 0) wred[wid] = ss;
    __syncthreads();
    if (tid == 0) {
        float t = 0.f;
        #pragma unroll
        for (int i = 0; i < 8; ++i) t += wred[i];
        sh_inv = rsqrtf(t * (1.0f / (float)NDF) + EPSF);
    }
    __syncthreads();
    const float inv = sh_inv;

    // ---- x_norm for this thread's K-span ------------------------------
    float xn[16];
    {
        const float4* rwv = (const float4*)(rms_w + k0);
        const float4* rbv = (const float4*)(rms_b + k0);
        #pragma unroll
        for (int q = 0; q < 4; ++q) {
            float4 rw = rwv[q], rb = rbv[q];
            xn[q*4+0] = xr[q*4+0] * inv * rw.x + rb.x;
            xn[q*4+1] = xr[q*4+1] * inv * rw.y + rb.y;
            xn[q*4+2] = xr[q*4+2] * inv * rw.z + rb.z;
            xn[q*4+3] = xr[q*4+3] * inv * rw.w + rb.w;
        }
    }

    // ---- 24 skinny dot products (w_pre[0..3], w_post[0..3], w_res[0..15])
    #pragma unroll
    for (int r = 0; r < 24; ++r) {
        const float* wr = (r < 4)  ? (w_pre  + (size_t)r       * NDF)
                        : (r < 8)  ? (w_post + (size_t)(r - 4) * NDF)
                                   : (w_res  + (size_t)(r - 8) * NDF);
        const float4* wv = (const float4*)(wr + k0);
        float acc = 0.f;
        #pragma unroll
        for (int q = 0; q < 4; ++q) {
            float4 w4 = wv[q];
            acc += xn[q*4+0] * w4.x + xn[q*4+1] * w4.y
                 + xn[q*4+2] * w4.z + xn[q*4+3] * w4.w;
        }
        acc = waveReduce(acc);
        if (lane == 0) dred[r][wid] = acc;
    }
    __syncthreads();
    if (tid < 24) {
        float t = 0.f;
        #pragma unroll
        for (int w = 0; w < 8; ++w) t += dred[tid][w];
        sh_dot[tid] = t;
    }
    __syncthreads();

    // ---- gates + Sinkhorn (serial on thread 0: 4x4 matrix, trivial) ----
    if (tid == 0) {
        const float apre  = *alpha_pre;
        const float apost = *alpha_post;
        const float ares  = *alpha_res;
        #pragma unroll
        for (int n = 0; n < 4; ++n) {
            float zp = apre * sh_dot[n] + b_pre[n];
            sh_hpre[n] = 1.0f / (1.0f + expf(-zp));
            float zq = apost * sh_dot[4 + n] + b_post[n];
            hpost_ws[(size_t)s * 4 + n] = 2.0f / (1.0f + expf(-zq));
        }
        float m[4][4];
        #pragma unroll
        for (int i = 0; i < 4; ++i)
            #pragma unroll
            for (int j = 0; j < 4; ++j)
                m[i][j] = expf(ares * sh_dot[8 + i * 4 + j] + b_res[i * 4 + j]);
        for (int it = 0; it < SKIT; ++it) {
            float cs0 = m[0][0]+m[1][0]+m[2][0]+m[3][0];
            float cs1 = m[0][1]+m[1][1]+m[2][1]+m[3][1];
            float cs2 = m[0][2]+m[1][2]+m[2][2]+m[3][2];
            float cs3 = m[0][3]+m[1][3]+m[2][3]+m[3][3];
            #pragma unroll
            for (int i = 0; i < 4; ++i) {
                m[i][0] /= cs0; m[i][1] /= cs1; m[i][2] /= cs2; m[i][3] /= cs3;
                float rs = m[i][0]+m[i][1]+m[i][2]+m[i][3];
                m[i][0] /= rs; m[i][1] /= rs; m[i][2] /= rs; m[i][3] /= rs;
            }
        }
        #pragma unroll
        for (int i = 0; i < 4; ++i)
            #pragma unroll
            for (int j = 0; j < 4; ++j)
                hres_ws[(size_t)s * 16 + i * 4 + j] = m[i][j];
    }
    __syncthreads();

    // ---- layer_input[s, d] = sum_n h_pre[n] * x[s, n, d]  (bf16 store) --
    const float hp0 = sh_hpre[0], hp1 = sh_hpre[1], hp2 = sh_hpre[2], hp3 = sh_hpre[3];
    #pragma unroll
    for (int j = 0; j < 4; ++j) {
        int d = tid + j * 256;
        float li = hp0 * xrow[d]
                 + hp1 * xrow[1024 + d]
                 + hp2 * xrow[2048 + d]
                 + hp3 * xrow[3072 + d];
        A_bf[((size_t)s << 10) + d] = f2bf(li);
    }
}

// ---------------------------------------------------------------------------
// Kernel 2: convert w_layer (1024x1024 f32) -> bf16
// ---------------------------------------------------------------------------
__global__ __launch_bounds__(256) void convw_kernel(
    const float* __restrict__ w, unsigned short* __restrict__ o)
{
    int i = blockIdx.x * 256 + threadIdx.x;   // grid = 4096 blocks
    o[i] = f2bf(w[i]);
}

// ---------------------------------------------------------------------------
// Kernel 3: layer_out = A_bf @ B_bf^T (TN, both K-contiguous) via WMMA bf16,
//           double-buffered LDS staging through the CDNA5 async-to-LDS path
//           (GLOBAL_LOAD_ASYNC_TO_LDS_B128 + s_wait_asynccnt) when available,
//           fused redistribute epilogue:
//   out[s,i,d] = sum_j hres[s,i,j]*x[s,j,d] + hpost[s,i]*layer_out[s,d]
// Block tile 128(M) x 128(N), K-slab 64, 8 waves of 2x4 16x16 WMMA tiles.
// ---------------------------------------------------------------------------
#define LDS_STRIDE 72   // 64 bf16 + 8 pad (16B) per row

__device__ __forceinline__ void cp16_g2l(const unsigned short* g, unsigned short* l) {
#if HAVE_ASYNC_LDS
    __builtin_amdgcn_global_load_async_to_lds_b128(
        (__attribute__((address_space(1))) v4i*)g,
        (__attribute__((address_space(3))) v4i*)l,
        0, 0);
#else
    *(uint4*)l = *(const uint4*)g;
#endif
}

__global__ __launch_bounds__(256) void gemm_epi_kernel(
    const unsigned short* __restrict__ A,   // (8192,1024) bf16 layer_input
    const unsigned short* __restrict__ B,   // (1024,1024) bf16 w_layer
    const float* __restrict__ x,            // (8192,4,1024)
    const float* __restrict__ hpost,        // (8192,4)
    const float* __restrict__ hres,         // (8192,16)
    float* __restrict__ out)                // (8192,4,1024)
{
    __shared__ __align__(16) unsigned short lA[2][128 * LDS_STRIDE];
    __shared__ __align__(16) unsigned short lB[2][128 * LDS_STRIDE];

    const int tid   = threadIdx.x;
    const int lane  = tid & 31;
    const int wid   = tid >> 5;
    const int waveM = wid & 3;     // 4 waves along M -> 32 rows each
    const int waveN = wid >> 2;    // 2 waves along N -> 64 cols each
    const int m0    = blockIdx.y * 128;
    const int n0    = blockIdx.x * 128;

    v8f acc[2][4] = {};

    // stage one 64-wide K-slab of A and B into LDS buffer `buf`
    auto stage = [&](int buf, int k0) {
        #pragma unroll
        for (int it = 0; it < 4; ++it) {
            int chunk = tid + it * 256;       // 1024 chunks of 16B per matrix
            int row   = chunk >> 3;
            int c8    = (chunk & 7) * 8;      // bf16 offset within row
            cp16_g2l(A + (((size_t)(m0 + row)) << 10) + k0 + c8,
                     &lA[buf][row * LDS_STRIDE + c8]);
            cp16_g2l(B + (((size_t)(n0 + row)) << 10) + k0 + c8,
                     &lB[buf][row * LDS_STRIDE + c8]);
        }
    };

    stage(0, 0);                               // prologue: fill buffer 0

    for (int ks = 0; ks < 16; ++ks) {          // 16 slabs of K=64
        const int nxt = ks + 1;
        if (nxt < 16) {
            stage(nxt & 1, nxt * 64);          // overlap next slab's copy
#if HAVE_ASYNC_LDS
            __builtin_amdgcn_s_wait_asynccnt(8);   // drain slab ks, keep nxt in flight
#endif
        } else {
#if HAVE_ASYNC_LDS
            __builtin_amdgcn_s_wait_asynccnt(0);   // drain the final slab
#endif
        }
        __syncthreads();

        const unsigned short* As = &lA[ks & 1][0];
        const unsigned short* Bs = &lB[ks & 1][0];

        // ---- two WMMA K-steps of 32 over this 64-wide slab --------------
        #pragma unroll
        for (int kk = 0; kk < 64; kk += 32) {
            union U { uint4 u[2]; v16bf v; };
            v16bf af[2];
            #pragma unroll
            for (int tm = 0; tm < 2; ++tm) {
                // A 16x32 bf16 layout: lanes 0-15 rows, chunk0 K=0..7 (+kc),
                // chunk1 K=+16; lanes 16-31 offset by 8 in K.
                int row = waveM * 32 + tm * 16 + (lane & 15);
                int kc  = kk + ((lane >> 4) << 3);
                U ua;
                ua.u[0] = *(const uint4*)&As[row * LDS_STRIDE + kc];
                ua.u[1] = *(const uint4*)&As[row * LDS_STRIDE + kc + 16];
                af[tm] = ua.v;
            }
            v16bf bfrag[4];
            #pragma unroll
            for (int tn = 0; tn < 4; ++tn) {
                // B 32x16: lanes 0-15 cols K=0..7,16..23; lanes 16-31 K=8..15,24..31
                int col = waveN * 64 + tn * 16 + (lane & 15);
                int kb  = kk + ((lane >> 4) << 3);
                U ub;
                ub.u[0] = *(const uint4*)&Bs[col * LDS_STRIDE + kb];
                ub.u[1] = *(const uint4*)&Bs[col * LDS_STRIDE + kb + 16];
                bfrag[tn] = ub.v;
            }
            #pragma unroll
            for (int tm = 0; tm < 2; ++tm)
                #pragma unroll
                for (int tn = 0; tn < 4; ++tn)
                    acc[tm][tn] = __builtin_amdgcn_wmma_f32_16x16x32_bf16(
                        false, af[tm], false, bfrag[tn],
                        (short)0, acc[tm][tn], false, false);
        }
        __syncthreads();   // protect buffer (ks&1) before slab ks+2 overwrites it
    }

    // ---- fused epilogue: redistribute to 4 streams ----------------------
    const int mlane = (lane >> 4) << 3;   // +0 or +8 row within 16x16 tile
    const int nlane = lane & 15;
    #pragma unroll
    for (int tm = 0; tm < 2; ++tm) {
        #pragma unroll
        for (int tn = 0; tn < 4; ++tn) {
            int sBase = m0 + waveM * 32 + tm * 16;
            int dcol  = n0 + waveN * 64 + tn * 16 + nlane;
            #pragma unroll
            for (int r = 0; r < 8; ++r) {
                int s = sBase + r + mlane;
                float lo = acc[tm][tn][r];
                const float* xr = x + (((size_t)s) << 12) + dcol;
                float x0 = xr[0], x1 = xr[1024], x2 = xr[2048], x3 = xr[3072];
                const float* hr = hres  + (size_t)s * 16;
                const float* hp = hpost + (size_t)s * 4;
                float* o = out + (((size_t)s) << 12) + dcol;
                #pragma unroll
                for (int i = 0; i < 4; ++i) {
                    o[i << 10] = hr[i*4+0]*x0 + hr[i*4+1]*x1
                               + hr[i*4+2]*x2 + hr[i*4+3]*x3
                               + hp[i]*lo;
                }
            }
        }
    }
}

// ---------------------------------------------------------------------------
// launch
// ---------------------------------------------------------------------------
extern "C" void kernel_launch(void* const* d_in, const int* in_sizes, int n_in,
                              void* d_out, int out_size, void* d_ws, size_t ws_size,
                              hipStream_t stream) {
    (void)in_sizes; (void)n_in; (void)out_size; (void)ws_size;

    const float* x        = (const float*)d_in[0];
    const float* w_pre    = (const float*)d_in[1];
    const float* w_post   = (const float*)d_in[2];
    const float* w_res    = (const float*)d_in[3];
    const float* b_pre    = (const float*)d_in[4];
    const float* b_post   = (const float*)d_in[5];
    const float* b_res    = (const float*)d_in[6];
    const float* a_pre    = (const float*)d_in[7];
    const float* a_post   = (const float*)d_in[8];
    const float* a_res    = (const float*)d_in[9];
    const float* rms_w    = (const float*)d_in[10];
    const float* rms_b    = (const float*)d_in[11];
    const float* w_layer  = (const float*)d_in[12];
    float* out            = (float*)d_out;

    // workspace layout
    char* ws = (char*)d_ws;
    unsigned short* A_bf   = (unsigned short*)ws;                                     // 16 MB
    unsigned short* B_bf   = (unsigned short*)(ws + (size_t)SEQ * DDIM * 2);          // 2 MB
    float*          hpost  = (float*)(ws + (size_t)SEQ * DDIM * 2 + (size_t)DDIM * DDIM * 2);
    float*          hresid = hpost + (size_t)SEQ * 4;

    prep_kernel<<<SEQ, 256, 0, stream>>>(
        x, w_pre, w_post, w_res, b_pre, b_post, b_res,
        a_pre, a_post, a_res, rms_w, rms_b,
        A_bf, hpost, hresid);

    convw_kernel<<<(DDIM * DDIM) / 256, 256, 0, stream>>>(w_layer, B_bf);

    dim3 grid(DDIM / 128, SEQ / 128);   // (8, 64)
    gemm_epi_kernel<<<grid, 256, 0, stream>>>(A_bf, B_bf, x, hpost, hresid, out);
}
